// DirectMFCModel_86062554677943
// MI455X (gfx1250) — matrix operand order repaired and checked by reading, not compiled, not run on hardware
//
#include <hip/hip_runtime.h>

typedef _Float16 h16;
typedef __attribute__((ext_vector_type(16))) _Float16 v16h;
typedef __attribute__((ext_vector_type(8)))  _Float16 v8h;
typedef __attribute__((ext_vector_type(8)))  float    v8f;

#define T_STEPS 200
#define N_PART  131072
#define MATURITY 1.0f
#define SIGMA   0.5f
#define C_A     1.0f
#define C_X     0.1f
#define GAMMA_  0.2f
#define C_G     0.3f

// ---- workspace layout (byte offsets into d_ws) ----
#define WS_ASUM   0        // 200 f32  per-step sum of a
#define WS_XSUM   800      // 200 f32  per-step sum of x
#define WS_S      1600     // 1 f32
#define WS_W2A    2048     // 32768B : W2^T in WMMA *A*-operand order
#define WS_W3A    34816    // 32768B : W3^T in A-operand order
#define WS_W1A    67584    // 8192B  : [W1;b1]^T (K rows 0..2) in A-operand order
#define WS_B2S    75776    // 512B   : b2 f32 (natural order == [nt][g][r])
#define WS_B3S    76288    // 512B
#define WS_W4S    76800    // 512B
#define WS_END    77312

// ---- dynamic LDS (bytes), 256-thread block = 8 waves, 4 tiles/wave ----
// [0,32768)         sW2 (A-operand)
// [32768,65536)     sW3
// [65536,73728)     sW1
// [73728,75264)     sb2/sb3/sw4 (f32)
// [75264,214528)    h buffers: 8 waves x 4 tiles x 16 rows x 136 halves
// [214528,282112)   dw staging: 8 waves x 64 x 33 f32
#define SIM_LDS_BYTES 282112

static __device__ __forceinline__ v8f wmma16(v16h a, v16h b, v8f c) {
    return __builtin_amdgcn_wmma_f32_16x16x32_f16(false, a, false, b,
                                                  (short)0, c, false, false);
}
static __device__ __forceinline__ v16h ldW(const h16* p) {
    v8h lo = *(const v8h*)p;
    v8h hi = *(const v8h*)(p + 8);
    return __builtin_shufflevector(lo, hi, 0,1,2,3,4,5,6,7,8,9,10,11,12,13,14,15);
}
static __device__ __forceinline__ v16h ldH(const h16* hb, int m, int kc, int g) {
    const h16* p = hb + m * 136 + kc * 32 + 8 * g;
    v8h lo = *(const v8h*)p;
    v8h hi = *(const v8h*)(p + 16);
    return __builtin_shufflevector(lo, hi, 0,1,2,3,4,5,6,7,8,9,10,11,12,13,14,15);
}
// D fragment -> 8 consecutive j of one particle: relu (packed f16) + one b128 store
static __device__ __forceinline__ void stDT(h16* dst, v8f acc) {
    v8h o;
    #pragma unroll
    for (int r = 0; r < 8; ++r) o[r] = (h16)acc[r];
#if __has_builtin(__builtin_elementwise_max)
    v8h z;
    #pragma unroll
    for (int r = 0; r < 8; ++r) z[r] = (h16)0.0f;
    o = __builtin_elementwise_max(o, z);         // v_pk_max_num_f16 x4
#else
    #pragma unroll
    for (int r = 0; r < 8; ++r) o[r] = (o[r] > (h16)0.0f) ? o[r] : (h16)0.0f;
#endif
    *(v8h*)dst = o;
}

// ============================================================
// Prep: zero accumulators; swizzle W^T matrices into WMMA
// A-operand order.  A element (M=j, K=k), block (kc, jt):
//   gA = (kl>>3)&1 ; hh = (kl&7) + 8*((kl>>4)&1)
//   lane = (j&15) + 16*gA ; idx = (kc*8+jt)*512 + lane*16 + hh
// ============================================================
__global__ void mfc_prep(const float* __restrict__ W1, const float* __restrict__ b1,
                         const float* __restrict__ W2, const float* __restrict__ b2,
                         const float* __restrict__ W3, const float* __restrict__ b3,
                         const float* __restrict__ W4, char* __restrict__ ws)
{
    const int tid = blockIdx.x * blockDim.x + threadIdx.x;   // 0..16383
    float* acc = (float*)(ws + WS_ASUM);
    if (tid < 401) acc[tid] = 0.0f;            // A_sum, X_sum, S

    {
        h16* w2a = (h16*)(ws + WS_W2A);
        h16* w3a = (h16*)(ws + WS_W3A);
        const int k = tid >> 7, j = tid & 127;
        const int kc = k >> 5, kl = k & 31;
        const int jt = j >> 4;
        const int gA = (kl >> 3) & 1;
        const int hh = (kl & 7) + ((kl >> 4) & 1) * 8;
        const int idx = (kc * 8 + jt) * 512 + ((j & 15) + 16 * gA) * 16 + hh;
        w2a[idx] = (h16)W2[tid];
        w3a[idx] = (h16)W3[tid];
    }

    if (tid < 4096) {   // [W1; b1]^T : K rows 0,1 = W1 ; row 2 = b1 ; rest 0
        h16* w1a = (h16*)(ws + WS_W1A);
        const int jt = tid >> 9;
        const int lane = (tid >> 4) & 31;
        const int hh = tid & 15;
        const int gA = lane >> 4;
        const int kl = (hh & 7) + 8 * gA + 16 * (hh >> 3);
        const int j = jt * 16 + (lane & 15);
        float v = 0.0f;
        if (kl == 0) v = W1[j];
        else if (kl == 1) v = W1[128 + j];
        else if (kl == 2) v = b1[j];
        w1a[tid] = (h16)v;
    }

    if (tid < 128) {   // natural order already matches [nt][g][r]
        ((float*)(ws + WS_B2S))[tid] = b2[tid];
        ((float*)(ws + WS_B3S))[tid] = b3[tid];
        ((float*)(ws + WS_W4S))[tid] = W4[tid];
    }
}

// ============================================================
// Simulation: 256 threads = 8 waves, 64 particles per wave
// (four 16-particle tiles sharing every weight A-fragment).
// ============================================================
__global__ __launch_bounds__(256, 1)
void mfc_sim(const float* __restrict__ xin, const float* __restrict__ dwg,
             const float* __restrict__ b4g, char* __restrict__ ws)
{
    extern __shared__ char smem[];
    h16*   sW2  = (h16*)smem;
    h16*   sW3  = (h16*)(smem + 32768);
    h16*   sW1  = (h16*)(smem + 65536);
    float* sb2  = (float*)(smem + 73728);
    float* sb3  = (float*)(smem + 74240);
    float* sw4  = (float*)(smem + 74752);
    h16*   hbuf = (h16*)(smem + 75264);
    float* sdw  = (float*)(smem + 214528);

    // stage all prepped weights/biases (ws[2048..77312) -> smem[0..75264))
    {
        const uint4* src = (const uint4*)(ws + WS_W2A);
        uint4* dst = (uint4*)smem;
        for (int i = threadIdx.x; i < 75264 / 16; i += 256) dst[i] = src[i];
    }
    __syncthreads();

    const int lane = threadIdx.x & 31;
    const int wv   = threadIdx.x >> 5;
    const int m    = lane & 15;     // particle within tile
    const int g    = lane >> 4;     // half-group
    const int pbase = blockIdx.x * 512 + wv * 64;
    const float dt = MATURITY / (float)T_STEPS;

    h16* hb[4];
    h16* st[4];
    #pragma unroll
    for (int tl = 0; tl < 4; ++tl) {
        hb[tl] = hbuf + wv * 8704 + tl * 2176;        // 16 x 136 halves
        st[tl] = hb[tl] + m * 136 + 8 * g;            // j-run store base
    }
    float* sdww = sdw + wv * 2112;                     // 64 x 33 f32

    float* Asum = (float*)(ws + WS_ASUM);
    float* Xsum = (float*)(ws + WS_XSUM);
    float* Stot = (float*)(ws + WS_S);

    const float b4v = b4g[0];
    float xp[4];
    #pragma unroll
    for (int tl = 0; tl < 4; ++tl) xp[tl] = xin[pbase + 16 * tl + m];
    float cost = 0.0f;

    #pragma unroll 1
    for (int t0 = 0; t0 < T_STEPS; t0 += 32) {
        #pragma unroll
        for (int p = 0; p < 64; ++p) {
            const int tt = t0 + lane;
            sdww[p * 33 + lane] =
                (tt < T_STEPS) ? dwg[(pbase + p) * T_STEPS + tt] : 0.0f;
        }
        const int tcnt = (T_STEPS - t0) < 32 ? (T_STEPS - t0) : 32;

        #pragma unroll 1
        for (int tc = 0; tc < tcnt; ++tc) {
            const int step = t0 + tc;
            const float tval = (float)step * dt;

            // ---- layer 1: B = per-particle [t, x, 1, 0...] column ----
            v16h bop[4];
            {
                v16h tmpl;
                #pragma unroll
                for (int hh = 0; hh < 16; ++hh) tmpl[hh] = (h16)0.0f;
                tmpl[0] = (h16)(g ? 0.0f : tval);
                tmpl[2] = (h16)(g ? 0.0f : 1.0f);
                #pragma unroll
                for (int tl = 0; tl < 4; ++tl) {
                    bop[tl] = tmpl;
                    bop[tl][1] = (h16)(g ? 0.0f : xp[tl]);
                }
            }

            v8f czero = {0.f, 0.f, 0.f, 0.f, 0.f, 0.f, 0.f, 0.f};
            #pragma unroll
            for (int nt = 0; nt < 8; ++nt) {
                v16h wf = ldW(&sW1[nt * 512 + lane * 16]);
                #pragma unroll
                for (int tl = 0; tl < 4; ++tl) {
                    v8f acc = wmma16(wf, bop[tl], czero);
                    stDT(st[tl] + nt * 16, acc);
                }
            }
            v16h bf[4][4];
            #pragma unroll
            for (int kc = 0; kc < 4; ++kc)
                #pragma unroll
                for (int tl = 0; tl < 4; ++tl)
                    bf[tl][kc] = ldH(hb[tl], m, kc, g);

            // ---- layer 2 ----
            #pragma unroll
            for (int nt = 0; nt < 8; ++nt) {
                v8f accs[4];
                const v8f binit = *(const v8f*)&sb2[nt * 16 + 8 * g];
                #pragma unroll
                for (int tl = 0; tl < 4; ++tl) accs[tl] = binit;
                #pragma unroll
                for (int kc = 0; kc < 4; ++kc) {
                    v16h wf = ldW(&sW2[(kc * 8 + nt) * 512 + lane * 16]);
                    #pragma unroll
                    for (int tl = 0; tl < 4; ++tl)
                        accs[tl] = wmma16(wf, bf[tl][kc], accs[tl]);
                }
                #pragma unroll
                for (int tl = 0; tl < 4; ++tl)
                    stDT(st[tl] + nt * 16, accs[tl]);
            }
            #pragma unroll
            for (int kc = 0; kc < 4; ++kc)
                #pragma unroll
                for (int tl = 0; tl < 4; ++tl)
                    bf[tl][kc] = ldH(hb[tl], m, kc, g);

            // ---- layer 3 + fused layer 4 (per-lane dot, no LDS) ----
            float pacc[4] = {0.0f, 0.0f, 0.0f, 0.0f};
            #pragma unroll
            for (int nt = 0; nt < 8; ++nt) {
                v8f accs[4];
                const v8f binit = *(const v8f*)&sb3[nt * 16 + 8 * g];
                #pragma unroll
                for (int tl = 0; tl < 4; ++tl) accs[tl] = binit;
                #pragma unroll
                for (int kc = 0; kc < 4; ++kc) {
                    v16h wf = ldW(&sW3[(kc * 8 + nt) * 512 + lane * 16]);
                    #pragma unroll
                    for (int tl = 0; tl < 4; ++tl)
                        accs[tl] = wmma16(wf, bf[tl][kc], accs[tl]);
                }
                const v8f w4v = *(const v8f*)&sw4[nt * 16 + 8 * g];
                #pragma unroll
                for (int r = 0; r < 8; ++r)
                    #pragma unroll
                    for (int tl = 0; tl < 4; ++tl)
                        pacc[tl] = fmaf(fmaxf(accs[tl][r], 0.0f), w4v[r], pacc[tl]);
            }
            float av[4];
            #pragma unroll
            for (int tl = 0; tl < 4; ++tl)
                av[tl] = pacc[tl] + __shfl_xor(pacc[tl], 16, 32) + b4v;

            // decoupled running cost (x before update); dt applied at end
            float aa = 0.0f, xx = 0.0f, sa = 0.0f, sx = 0.0f;
            #pragma unroll
            for (int tl = 0; tl < 4; ++tl) {
                aa += av[tl] * av[tl];
                xx += xp[tl] * xp[tl];
                sa += av[tl];
                sx += xp[tl];
            }
            cost = fmaf(0.5f * C_A, aa, cost);
            cost = fmaf(0.5f * C_X, xx, cost);

            // per-step global sums for the mean-field cross term
            #pragma unroll
            for (int off = 16; off > 0; off >>= 1) {
                sa += __shfl_xor(sa, off, 32);
                sx += __shfl_xor(sx, off, 32);
            }
            if (lane == 0) {                    // halves duplicate => *0.5
                atomicAdd(&Asum[step], sa * 0.5f);
                atomicAdd(&Xsum[step], sx * 0.5f);
            }

            // dynamics
            #pragma unroll
            for (int tl = 0; tl < 4; ++tl)
                xp[tl] = fmaf(av[tl], dt,
                              fmaf(SIGMA, sdww[(m + 16 * tl) * 33 + tc], xp[tl]));
        }
    }

    float xxT = 0.0f;
    #pragma unroll
    for (int tl = 0; tl < 4; ++tl) xxT += xp[tl] * xp[tl];
    cost = cost * dt + 0.5f * C_G * xxT;
    #pragma unroll
    for (int off = 16; off > 0; off >>= 1) cost += __shfl_xor(cost, off, 32);
    if (lane == 0) atomicAdd(Stot, cost * 0.5f);
}

// ============================================================
// Finalize: mean cost + Sum_t gamma*dt*(A[t]/N)*(X[t]/N)
// ============================================================
__global__ void mfc_final(const char* __restrict__ ws, float* __restrict__ out)
{
    __shared__ float red[256];
    const float* Asum = (const float*)(ws + WS_ASUM);
    const float* Xsum = (const float*)(ws + WS_XSUM);
    const float* Stot = (const float*)(ws + WS_S);
    float p = 0.0f;
    for (int t = threadIdx.x; t < T_STEPS; t += 256)
        p += Asum[t] * Xsum[t];
    red[threadIdx.x] = p;
    __syncthreads();
    for (int s = 128; s > 0; s >>= 1) {
        if (threadIdx.x < s) red[threadIdx.x] += red[threadIdx.x + s];
        __syncthreads();
    }
    if (threadIdx.x == 0) {
        const float invN = 1.0f / (float)N_PART;
        const float dt = MATURITY / (float)T_STEPS;
        out[0] = Stot[0] * invN + GAMMA_ * dt * red[0] * invN * invN;
    }
}

extern "C" void kernel_launch(void* const* d_in, const int* in_sizes, int n_in,
                              void* d_out, int out_size, void* d_ws, size_t ws_size,
                              hipStream_t stream)
{
    const float* x  = (const float*)d_in[0];
    const float* dw = (const float*)d_in[1];
    const float* W1 = (const float*)d_in[2];
    const float* b1 = (const float*)d_in[3];
    const float* W2 = (const float*)d_in[4];
    const float* b2 = (const float*)d_in[5];
    const float* W3 = (const float*)d_in[6];
    const float* b3 = (const float*)d_in[7];
    const float* W4 = (const float*)d_in[8];
    const float* b4 = (const float*)d_in[9];
    char* ws = (char*)d_ws;
    (void)in_sizes; (void)n_in; (void)out_size; (void)ws_size;

    mfc_prep<<<64, 256, 0, stream>>>(W1, b1, W2, b2, W3, b3, W4, ws);
    mfc_sim<<<N_PART / 512, 256, SIM_LDS_BYTES, stream>>>(x, dw, b4, ws);
    mfc_final<<<1, 256, 0, stream>>>(ws, (float*)d_out);
}
